// ScaledDotProductAttention_21784074125624
// MI455X (gfx1250) — matrix operand order repaired and checked by reading, not compiled, not run on hardware
//
#include <hip/hip_runtime.h>
#include <hip/hip_bf16.h>
#include <stdint.h>

// Problem constants (from reference): B=2, H=16, S=2048, D=64
#define SB 2
#define SH 16
#define SS 2048
#define SD 64

typedef __attribute__((ext_vector_type(16))) __bf16 v16bf;
typedef __attribute__((ext_vector_type(8)))  float  v8f;

union FragA {
    v16bf bf;
    uint4 q[2];
    unsigned int u[8];
};

__device__ __forceinline__ unsigned short f2bf(float f) {
    union { float f; unsigned int u; } x;
    x.f = f;
    unsigned int r = x.u + 0x7FFFu + ((x.u >> 16) & 1u);  // round-to-nearest-even
    return (unsigned short)(r >> 16);
}

__device__ __forceinline__ uint2 pack4bf(float4 v) {
    uint2 p;
    p.x = (unsigned)f2bf(v.x) | ((unsigned)f2bf(v.y) << 16);
    p.y = (unsigned)f2bf(v.z) | ((unsigned)f2bf(v.w) << 16);
    return p;
}

// max-reduce across a 16-lane DPP row (wave32: halves stay independent)
__device__ __forceinline__ float rowmax16(float v) {
    int x;
    x = __builtin_amdgcn_update_dpp(0, __float_as_int(v), 0x121, 0xf, 0xf, true); // row_ror:1
    v = fmaxf(v, __int_as_float(x));
    x = __builtin_amdgcn_update_dpp(0, __float_as_int(v), 0x122, 0xf, 0xf, true); // row_ror:2
    v = fmaxf(v, __int_as_float(x));
    x = __builtin_amdgcn_update_dpp(0, __float_as_int(v), 0x124, 0xf, 0xf, true); // row_ror:4
    v = fmaxf(v, __int_as_float(x));
    x = __builtin_amdgcn_update_dpp(0, __float_as_int(v), 0x128, 0xf, 0xf, true); // row_ror:8
    v = fmaxf(v, __int_as_float(x));
    return v;
}

__device__ __forceinline__ void wait_async0() {
#if __has_builtin(__builtin_amdgcn_s_wait_asynccnt)
    __builtin_amdgcn_s_wait_asynccnt(0);
#else
    asm volatile("s_wait_asynccnt 0x0" ::: "memory");
#endif
}

// issue one per-lane 16B async copy global->LDS (ASYNCcnt-tracked)
__device__ __forceinline__ void async_b128(unsigned lds_byte_addr, unsigned long long gaddr) {
    asm volatile("global_load_async_to_lds_b128 %0, %1, off"
                 :: "v"(lds_byte_addr), "v"(gaddr) : "memory");
}

// ======================= prepass: fp32 -> bf16 (K) and bf16 transposed (V) ====
// grid: (S/64, B*H), 256 threads. Kb: [bh][s][d] bf16. Vtb: [bh][d][s] bf16.
__global__ __launch_bounds__(256) void prepass_kernel(
    const float* __restrict__ K, const float* __restrict__ V,
    unsigned short* __restrict__ Kb, unsigned short* __restrict__ Vtb)
{
    __shared__ unsigned short sT[64 * 64];
    const int tid = threadIdx.x;
    const int bh  = blockIdx.y;
    const int s0  = blockIdx.x * 64;
    const size_t base = (size_t)bh * SS * SD;

    for (int i = 0; i < 4; ++i) {
        int lin = i * 256 + tid;          // 1024 float4
        int row = lin >> 4, c4 = lin & 15;
        const size_t off = base + (size_t)(s0 + row) * SD + c4 * 4;
        *(uint2*)&Kb[off] = pack4bf(*(const float4*)(K + off));
        *(uint2*)&sT[row * 64 + c4 * 4] = pack4bf(*(const float4*)(V + off));
    }
    __syncthreads();
    for (int i = 0; i < 4; ++i) {
        int lin = i * 256 + tid;          // 1024 x 8B chunks of Vt
        int d = lin >> 3, c = lin & 7;    // 8 chunks x 4 elems = 32 s per d row
        int s = c * 4;
        uint2 o;
        o.x = (unsigned)sT[(s + 0) * 64 + d] | ((unsigned)sT[(s + 1) * 64 + d] << 16);
        o.y = (unsigned)sT[(s + 2) * 64 + d] | ((unsigned)sT[(s + 3) * 64 + d] << 16);
        *(uint2*)&Vtb[((size_t)bh * SD + d) * SS + s0 + s] = o;
    }
}

// ======================= fast path: async double-buffered flash attention ====
__global__ __launch_bounds__(128) void fa_wmma_kernel_async(
    const float* __restrict__ Q,
    const unsigned short* __restrict__ Kb,   // bf16 [bh][s][d]
    const unsigned short* __restrict__ Vtb,  // bf16 [bh][d][s]
    float* __restrict__ O)
{
    __shared__ unsigned short sQ[64 * 64];          // 8 KB
    __shared__ unsigned short sK[2][32 * 64];       // 2 x 4 KB
    __shared__ unsigned short sVt[2][64 * 32];      // 2 x 4 KB
    __shared__ unsigned short sP[4][16 * 32];       // 4 KB

    const int tid  = threadIdx.x;
    const int wave = tid >> 5;
    const int lane = tid & 31;
    const int half = lane >> 4;
    const int l16  = lane & 15;

    const int bh = blockIdx.y;
    const int q0 = blockIdx.x * 64;
    const size_t base = (size_t)bh * SS * SD;
    const float scale = 0.125f;
    const int NT = SS / 32;

    // stage Q (scaled) into LDS as bf16
    for (int i = 0; i < 8; ++i) {
        int lin = i * 128 + tid;
        int row = lin >> 4, c4 = lin & 15;
        float4 qv = *(const float4*)(Q + base + (size_t)(q0 + row) * SD + c4 * 4);
        qv.x *= scale; qv.y *= scale; qv.z *= scale; qv.w *= scale;
        *(uint2*)&sQ[row * 64 + c4 * 4] = pack4bf(qv);
    }

    const unsigned ldsK0 = (unsigned)(uintptr_t)&sK[0][0];
    const unsigned ldsV0 = (unsigned)(uintptr_t)&sVt[0][0];

    // async stage of K/V tile t into buffer b (4 async b128 per wave)
    auto stage = [&](int t, int b) {
        const unsigned long long kB =
            (unsigned long long)(uintptr_t)Kb + (base + (size_t)t * 32 * SD) * 2;
        for (int i = 0; i < 2; ++i) {
            int c = i * 128 + tid;                  // 256 x 16B = 4 KB, contiguous
            async_b128(ldsK0 + (unsigned)(b * 4096 + c * 16), kB + (unsigned)c * 16);
        }
        const unsigned long long vB =
            (unsigned long long)(uintptr_t)Vtb + ((size_t)bh * SD * SS + (size_t)t * 32) * 2;
        for (int i = 0; i < 2; ++i) {
            int c = i * 128 + tid;                  // 256 chunks: 64 rows x 4
            int d = c >> 2, sub = c & 3;
            async_b128(ldsV0 + (unsigned)(b * 4096 + d * 64 + sub * 16),
                       vB + (size_t)d * SS * 2 + (unsigned)(sub * 16));
        }
    };

    stage(0, 0);
    __syncthreads();

    // per-wave Q A-fragments
    FragA aq[2];
    {
        const unsigned short* qrow = &sQ[(wave * 16 + l16) * 64];
        for (int f = 0; f < 2; ++f) {
            int k0 = f * 32 + half * 8;
            aq[f].q[0] = *(const uint4*)(qrow + k0);
            aq[f].q[1] = *(const uint4*)(qrow + k0 + 16);
        }
    }
    FragA ones;
    for (int i = 0; i < 8; ++i) ones.u[i] = 0x3F803F80u;

    v8f acc[4], lacc;
    for (int n = 0; n < 4; ++n)
        for (int j = 0; j < 8; ++j) acc[n][j] = 0.0f;
    for (int j = 0; j < 8; ++j) lacc[j] = 0.0f;
    float mrun[8];
    for (int j = 0; j < 8; ++j) mrun[j] = -1e30f;

    for (int t = 0; t < NT; ++t) {
        wait_async0();                    // this wave's stage(t) copies landed
        __syncthreads();                  // everyone's landed; old buffer free
        if (t + 1 < NT) stage(t + 1, (t + 1) & 1);
        const unsigned short* bK  = sK[t & 1];
        const unsigned short* bVt = sVt[t & 1];

        // batch all four K B-fragments, then run the WMMA chain
        FragA bk[4];                      // [nt*2+f]
        for (int nt = 0; nt < 2; ++nt)
            for (int f = 0; f < 2; ++f) {
                const unsigned short* kr = &bK[(nt * 16 + l16) * 64 + f * 32 + half * 8];
                bk[nt * 2 + f].q[0] = *(const uint4*)(kr);
                bk[nt * 2 + f].q[1] = *(const uint4*)(kr + 16);
            }
        v8f c01[2];
        for (int nt = 0; nt < 2; ++nt) {
            v8f c = {0.f, 0.f, 0.f, 0.f, 0.f, 0.f, 0.f, 0.f};
            c = __builtin_amdgcn_wmma_f32_16x16x32_bf16(
                    false, aq[0].bf, false, bk[nt * 2].bf, (short)0, c, false, false);
            c = __builtin_amdgcn_wmma_f32_16x16x32_bf16(
                    false, aq[1].bf, false, bk[nt * 2 + 1].bf, (short)0, c, false, false);
            c01[nt] = c;
        }

        // online softmax
        float mnew[8], fac[8];
        for (int j = 0; j < 8; ++j) {
            float mx = rowmax16(fmaxf(c01[0][j], c01[1][j]));
            mnew[j] = fmaxf(mrun[j], mx);
            fac[j]  = __expf(mrun[j] - mnew[j]);
            mrun[j] = mnew[j];
        }
        unsigned short* pw = sP[wave];
        for (int j = 0; j < 8; ++j) {
            float p0 = __expf(c01[0][j] - mnew[j]);
            float p1 = __expf(c01[1][j] - mnew[j]);
            int row = j + half * 8;
            pw[row * 32 + l16]      = f2bf(p0);
            pw[row * 32 + 16 + l16] = f2bf(p1);
        }
        for (int n = 0; n < 4; ++n)
            for (int j = 0; j < 8; ++j) acc[n][j] *= fac[j];
        for (int j = 0; j < 8; ++j) lacc[j] *= fac[j];

        asm volatile("" ::: "memory");   // same-wave LDS RAW (HW keeps DS in order)

        FragA ap;
        {
            const unsigned short* pr = &pw[l16 * 32 + half * 8];
            ap.q[0] = *(const uint4*)(pr);
            ap.q[1] = *(const uint4*)(pr + 16);
        }
        FragA bv[4];
        for (int n = 0; n < 4; ++n) {
            const unsigned short* vr = &bVt[(n * 16 + l16) * 32 + half * 8];
            bv[n].q[0] = *(const uint4*)(vr);
            bv[n].q[1] = *(const uint4*)(vr + 16);
        }
        lacc = __builtin_amdgcn_wmma_f32_16x16x32_bf16(
                   false, ap.bf, false, ones.bf, (short)0, lacc, false, false);
        for (int n = 0; n < 4; ++n)
            acc[n] = __builtin_amdgcn_wmma_f32_16x16x32_bf16(
                         false, ap.bf, false, bv[n].bf, (short)0, acc[n], false, false);
    }

    for (int j = 0; j < 8; ++j) {
        float inv = 1.0f / lacc[j];
        int row = q0 + wave * 16 + j + half * 8;
        float* orow = O + base + (size_t)row * SD;
        for (int n = 0; n < 4; ++n)
            orow[n * 16 + l16] = acc[n][j] * inv;
    }
}

// ======================= fallback: in-kernel conversion (round-2 path) =======
__global__ __launch_bounds__(128) void fa_wmma_kernel_conv(
    const float* __restrict__ Q, const float* __restrict__ K,
    const float* __restrict__ V, float* __restrict__ O)
{
    __shared__ unsigned short sQ[64 * 64];
    __shared__ unsigned short sK[2][32 * 64];
    __shared__ unsigned short sVt[2][64 * 32];
    __shared__ unsigned short sP[4][16 * 32];

    const int tid  = threadIdx.x;
    const int wave = tid >> 5;
    const int lane = tid & 31;
    const int half = lane >> 4;
    const int l16  = lane & 15;

    const int bh = blockIdx.y;
    const int q0 = blockIdx.x * 64;
    const size_t base = (size_t)bh * SS * SD;
    const float scale = 0.125f;
    const int NT = SS / 32;

    for (int i = 0; i < 8; ++i) {
        int lin = i * 128 + tid;
        int row = lin >> 4, c4 = lin & 15;
        float4 qv = *(const float4*)(Q + base + (size_t)(q0 + row) * SD + c4 * 4);
        qv.x *= scale; qv.y *= scale; qv.z *= scale; qv.w *= scale;
        *(uint2*)&sQ[row * 64 + c4 * 4] = pack4bf(qv);
    }

    auto stage = [&](int t, int b) {
        for (int i = 0; i < 4; ++i) {
            int lin = i * 128 + tid;
            int row = lin >> 4, c4 = lin & 15;
            const size_t goff = base + (size_t)(t * 32 + row) * SD + c4 * 4;
            *(uint2*)&sK[b][row * 64 + c4 * 4] = pack4bf(*(const float4*)(K + goff));
            const float4 vv = *(const float4*)(V + goff);
            sVt[b][(c4 * 4 + 0) * 32 + row] = f2bf(vv.x);
            sVt[b][(c4 * 4 + 1) * 32 + row] = f2bf(vv.y);
            sVt[b][(c4 * 4 + 2) * 32 + row] = f2bf(vv.z);
            sVt[b][(c4 * 4 + 3) * 32 + row] = f2bf(vv.w);
            if (t + 1 < NT) {
                __builtin_prefetch(K + goff + 32 * SD, 0, 1);
                __builtin_prefetch(V + goff + 32 * SD, 0, 1);
            }
        }
    };

    stage(0, 0);
    __syncthreads();

    FragA aq[2];
    {
        const unsigned short* qrow = &sQ[(wave * 16 + l16) * 64];
        for (int f = 0; f < 2; ++f) {
            int k0 = f * 32 + half * 8;
            aq[f].q[0] = *(const uint4*)(qrow + k0);
            aq[f].q[1] = *(const uint4*)(qrow + k0 + 16);
        }
    }
    FragA ones;
    for (int i = 0; i < 8; ++i) ones.u[i] = 0x3F803F80u;

    v8f acc[4], lacc;
    for (int n = 0; n < 4; ++n)
        for (int j = 0; j < 8; ++j) acc[n][j] = 0.0f;
    for (int j = 0; j < 8; ++j) lacc[j] = 0.0f;
    float mrun[8];
    for (int j = 0; j < 8; ++j) mrun[j] = -1e30f;

    for (int t = 0; t < NT; ++t) {
        __syncthreads();
        if (t + 1 < NT) stage(t + 1, (t + 1) & 1);
        const unsigned short* bK  = sK[t & 1];
        const unsigned short* bVt = sVt[t & 1];

        FragA bk[4];
        for (int nt = 0; nt < 2; ++nt)
            for (int f = 0; f < 2; ++f) {
                const unsigned short* kr = &bK[(nt * 16 + l16) * 64 + f * 32 + half * 8];
                bk[nt * 2 + f].q[0] = *(const uint4*)(kr);
                bk[nt * 2 + f].q[1] = *(const uint4*)(kr + 16);
            }
        v8f c01[2];
        for (int nt = 0; nt < 2; ++nt) {
            v8f c = {0.f, 0.f, 0.f, 0.f, 0.f, 0.f, 0.f, 0.f};
            c = __builtin_amdgcn_wmma_f32_16x16x32_bf16(
                    false, aq[0].bf, false, bk[nt * 2].bf, (short)0, c, false, false);
            c = __builtin_amdgcn_wmma_f32_16x16x32_bf16(
                    false, aq[1].bf, false, bk[nt * 2 + 1].bf, (short)0, c, false, false);
            c01[nt] = c;
        }

        float mnew[8], fac[8];
        for (int j = 0; j < 8; ++j) {
            float mx = rowmax16(fmaxf(c01[0][j], c01[1][j]));
            mnew[j] = fmaxf(mrun[j], mx);
            fac[j]  = __expf(mrun[j] - mnew[j]);
            mrun[j] = mnew[j];
        }
        unsigned short* pw = sP[wave];
        for (int j = 0; j < 8; ++j) {
            float p0 = __expf(c01[0][j] - mnew[j]);
            float p1 = __expf(c01[1][j] - mnew[j]);
            int row = j + half * 8;
            pw[row * 32 + l16]      = f2bf(p0);
            pw[row * 32 + 16 + l16] = f2bf(p1);
        }
        for (int n = 0; n < 4; ++n)
            for (int j = 0; j < 8; ++j) acc[n][j] *= fac[j];
        for (int j = 0; j < 8; ++j) lacc[j] *= fac[j];

        asm volatile("" ::: "memory");

        FragA ap;
        {
            const unsigned short* pr = &pw[l16 * 32 + half * 8];
            ap.q[0] = *(const uint4*)(pr);
            ap.q[1] = *(const uint4*)(pr + 16);
        }
        FragA bv[4];
        for (int n = 0; n < 4; ++n) {
            const unsigned short* vr = &bVt[(n * 16 + l16) * 32 + half * 8];
            bv[n].q[0] = *(const uint4*)(vr);
            bv[n].q[1] = *(const uint4*)(vr + 16);
        }
        lacc = __builtin_amdgcn_wmma_f32_16x16x32_bf16(
                   false, ap.bf, false, ones.bf, (short)0, lacc, false, false);
        for (int n = 0; n < 4; ++n)
            acc[n] = __builtin_amdgcn_wmma_f32_16x16x32_bf16(
                         false, ap.bf, false, bv[n].bf, (short)0, acc[n], false, false);
    }

    for (int j = 0; j < 8; ++j) {
        float inv = 1.0f / lacc[j];
        int row = q0 + wave * 16 + j + half * 8;
        float* orow = O + base + (size_t)row * SD;
        for (int n = 0; n < 4; ++n)
            orow[n * 16 + l16] = acc[n][j] * inv;
    }
}

extern "C" void kernel_launch(void* const* d_in, const int* in_sizes, int n_in,
                              void* d_out, int out_size, void* d_ws, size_t ws_size,
                              hipStream_t stream) {
    const float* q = (const float*)d_in[0];
    const float* k = (const float*)d_in[1];
    const float* v = (const float*)d_in[2];
    float* out = (float*)d_out;
    (void)in_sizes; (void)n_in; (void)out_size;

    const size_t elems = (size_t)SB * SH * SS * SD;      // 4M
    const size_t need  = elems * 2 * 2;                  // Kb + Vtb bf16 = 16 MB
    dim3 grid(SS / 64, SB * SH);
    if (d_ws && ws_size >= need) {
        unsigned short* Kb  = (unsigned short*)d_ws;
        unsigned short* Vtb = Kb + elems;
        prepass_kernel<<<grid, 256, 0, stream>>>(k, v, Kb, Vtb);
        fa_wmma_kernel_async<<<grid, 128, 0, stream>>>(q, Kb, Vtb, out);
    } else {
        fa_wmma_kernel_conv<<<grid, 128, 0, stream>>>(q, k, v, out);
    }
}
